// DMAGLSTMCell_19155554140195
// MI455X (gfx1250) — matrix operand order repaired and checked by reading, not compiled, not run on hardware
//
#include <hip/hip_runtime.h>
#include <math.h>

// Problem constants
#define B_   64
#define T_   512
#define D_   256
#define U_   512
#define K_   768            // U + D
#define NG   2304           // 4U + D (full gates width)
#define NSG  2048           // stored gate width (f_s,f_l,alpha,o) -- m stays in LDS
#define NT_G (NG / 16)      // 144 n-tiles for gates gemm
#define NT_C (U_ / 16)      // 32  n-tiles for cell gemm
#define KS   (K_ / 32)      // 24  k-steps of 32
#define KS_H 16             // k-steps covering h (k < 512)
#define KS_X 8              // k-steps covering x part (k >= 512)
#define AX_STEP (4 * KS_X * 32 * 8)   // dwords of x-fragments per timestep (8192)

typedef __attribute__((ext_vector_type(16))) __bf16       v16bf;
typedef __attribute__((ext_vector_type(8)))  float        v8f;
typedef __attribute__((ext_vector_type(8)))  unsigned int v8u;

__device__ __forceinline__ unsigned short f2bf(float f) {
    unsigned int u = __float_as_uint(f);
    unsigned int r = u + 0x7FFFu + ((u >> 16) & 1u);   // round-to-nearest-even
    return (unsigned short)(r >> 16);
}
__device__ __forceinline__ unsigned int pack2(float lo, float hi) {
    return (unsigned int)f2bf(lo) | ((unsigned int)f2bf(hi) << 16);
}
__device__ __forceinline__ float sigmoidf_(float x) { return 1.0f / (1.0f + __expf(-x)); }
// A-fragment k offset for VGPR v (16-bit A 16x32 layout)
__device__ __forceinline__ int aoff(int v) { return (v < 4) ? 2 * v : 16 + 2 * (v - 4); }

// ---------------------------------------------------------------------------
// Shared WMMA inner segment: nks k-steps; B pointers advance by reference so
// the h-part and x-part segments chain over one contiguous B stream.
// ---------------------------------------------------------------------------
__device__ __forceinline__ void mma_seg(const unsigned int* __restrict__ ap,
                                        const unsigned int*& bp0,
                                        const unsigned int*& bp1,
                                        int nks, v8f& acc0, v8f& acc1) {
    for (int ks = 0; ks < nks; ++ks) {
        v8u av = *(const v8u*)ap;
        v8u b0 = *(const v8u*)bp0;
        v8u b1 = *(const v8u*)bp1;
        __builtin_prefetch(bp0 + 256, 0, 1);   // global_prefetch_b8: next k-slice
        __builtin_prefetch(bp1 + 256, 0, 1);
        v16bf a = __builtin_bit_cast(v16bf, av);
        acc0 = __builtin_amdgcn_wmma_f32_16x16x32_bf16(
            false, a, false, __builtin_bit_cast(v16bf, b0), (short)0, acc0, false, false);
        acc1 = __builtin_amdgcn_wmma_f32_16x16x32_bf16(
            false, a, false, __builtin_bit_cast(v16bf, b1), (short)0, acc1, false, false);
        ap += 256; bp0 += 256; bp1 += 256;
    }
}

// ---------------------------------------------------------------------------
// init: copy c0 -> cbuf; pack h0 into Ah[0] fragments (ksteps 0..15)
// ---------------------------------------------------------------------------
__global__ void init_state(const float* __restrict__ h0, const float* __restrict__ c0,
                           float* __restrict__ cbuf, unsigned int* __restrict__ Ah0) {
    int i = blockIdx.x * blockDim.x + threadIdx.x;
    if (i < B_ * U_) cbuf[i] = c0[i];
    if (i < 4 * KS_H * 32) {
        int lane  = i & 31;
        int ks    = (i >> 5) % KS_H;
        int mtile = i / (32 * KS_H);
        int row   = mtile * 16 + (lane & 15);
        int khalf = (lane >> 4) ? 8 : 0;
        unsigned int* d = Ah0 + (((size_t)mtile * KS_H + ks) * 32 + lane) * 8;
#pragma unroll
        for (int v = 0; v < 8; ++v) {
            int k = ks * 32 + aoff(v) + khalf;
            d[v] = pack2(h0[row * U_ + k], h0[row * U_ + k + 1]);
        }
    }
}

// ---------------------------------------------------------------------------
// Precompute x-part A fragments for ALL timesteps (ksteps 16..23).
// ---------------------------------------------------------------------------
__global__ void pack_x_all(const float* __restrict__ x, unsigned int* __restrict__ AxX) {
    int idx = blockIdx.x * blockDim.x + threadIdx.x;   // 512*4*8*32 threads
    int lane  = idx & 31;
    int ks8   = (idx >> 5) & 7;
    int mtile = (idx >> 8) & 3;
    int t     = idx >> 10;
    if (t >= T_) return;
    int row   = mtile * 16 + (lane & 15);
    int khalf = (lane >> 4) ? 8 : 0;
    unsigned int* d = AxX + (size_t)t * AX_STEP + (((size_t)mtile * KS_X + ks8) * 32 + lane) * 8;
#pragma unroll
    for (int v = 0; v < 8; ++v) {
        int k = (KS_H + ks8) * 32 + aoff(v) + khalf;   // 512..767
        int dd = k - U_;
        const float* xr = x + (size_t)row * T_ * D_ + (size_t)t * D_;
        d[v] = pack2(xr[dd], xr[dd + 1]);
    }
}

// ---------------------------------------------------------------------------
// Weight packing: fp32 row-major -> bf16 WMMA B-fragments (contiguous/lane).
// ---------------------------------------------------------------------------
__device__ __forceinline__ float wg_at(const float* wfs, const float* wfl, const float* wal,
                                       const float* wo, const float* wm, int k, int n) {
    if (n < 512)  return wfs[k * 512 + n];
    if (n < 1024) return wfl[k * 512 + (n - 512)];
    if (n < 1536) return wal[k * 512 + (n - 1024)];
    if (n < 2048) return wo [k * 512 + (n - 1536)];
    return               wm [k * 256 + (n - 2048)];
}

__global__ void pack_weights(const float* __restrict__ wfs, const float* __restrict__ wfl,
                             const float* __restrict__ wal, const float* __restrict__ wo,
                             const float* __restrict__ wm,  const float* __restrict__ wC,
                             unsigned int* __restrict__ Bg, unsigned int* __restrict__ Bc) {
    int idx = blockIdx.x * blockDim.x + threadIdx.x;
    const int TOT = (NT_G + NT_C) * KS * 32;
    if (idx >= TOT) return;
    int lane  = idx & 31;
    int ks    = (idx >> 5) % KS;
    int nt    = idx / (32 * KS);
    int col   = lane & 15;
    int khalf = (lane >> 4) ? 8 : 0;
    if (nt < NT_G) {
        int n = nt * 16 + col;
        unsigned int* dst = Bg + (((size_t)nt * KS + ks) * 32 + lane) * 8;
#pragma unroll
        for (int v = 0; v < 8; ++v) {
            int k0 = ks * 32 + aoff(v) + khalf;
            dst[v] = pack2(wg_at(wfs, wfl, wal, wo, wm, k0, n),
                           wg_at(wfs, wfl, wal, wo, wm, k0 + 1, n));
        }
    } else {
        int ntc = nt - NT_G;
        int n = ntc * 16 + col;
        unsigned int* dst = Bc + (((size_t)ntc * KS + ks) * 32 + lane) * 8;
#pragma unroll
        for (int v = 0; v < 8; ++v) {
            int k0 = ks * 32 + aoff(v) + khalf;
            dst[v] = pack2(wC[k0 * 512 + n], wC[(k0 + 1) * 512 + n]);
        }
    }
}

__device__ __forceinline__ float gate_bias(const float* bfs, const float* bfl,
                                           const float* bal, const float* bo,
                                           const float* bm, int n) {
    if (n < 512)  return bfs[n];
    if (n < 1024) return bfl[n - 512];
    if (n < 1536) return bal[n - 1024];
    if (n < 2048) return bo [n - 1536];
    return               bm [n - 2048];
}

// ---------------------------------------------------------------------------
// GEMM gates: [64,768]x[768,2304]. Blocks 0..31 write sigmoid(gates) to sg.
// Blocks 32..35 (m-gate columns) keep sigmoid(m) in LDS and emit the m*x
// A-fragments (ksteps 16..23) for this step's cell gemm.
// ---------------------------------------------------------------------------
__global__ void gemm_gates(const unsigned int* __restrict__ Ah,
                           const unsigned int* __restrict__ AxXt,
                           const unsigned int* __restrict__ Bg,
                           const float* __restrict__ bfs, const float* __restrict__ bfl,
                           const float* __restrict__ bal, const float* __restrict__ bo,
                           const float* __restrict__ bm,
                           float* __restrict__ sg,
                           const float* __restrict__ x,
                           unsigned int* __restrict__ AmX, int t) {
    __shared__ float mlds[64][64];
    int lane   = threadIdx.x & 31;
    int wave   = threadIdx.x >> 5;
    int mtile  = wave & 3;
    int nl     = (wave >> 2) * 2;
    int ntile0 = blockIdx.x * 4 + nl;
    int ntile1 = ntile0 + 1;

    const unsigned int* ahp = Ah   + (((size_t)mtile * KS_H) * 32 + lane) * 8;
    const unsigned int* axp = AxXt + (((size_t)mtile * KS_X) * 32 + lane) * 8;
    const unsigned int* bp0 = Bg + (((size_t)ntile0 * KS) * 32 + lane) * 8;
    const unsigned int* bp1 = Bg + (((size_t)ntile1 * KS) * 32 + lane) * 8;

    v8f acc0 = {}; v8f acc1 = {};
    mma_seg(ahp, bp0, bp1, KS_H, acc0, acc1);   // h part (k<512)
    mma_seg(axp, bp0, bp1, KS_X, acc0, acc1);   // x part (k>=512)

    int col0    = ntile0 * 16 + (lane & 15);
    int col1    = ntile1 * 16 + (lane & 15);
    int rowbase = mtile * 16 + ((lane >> 4) << 3);
    float bb0 = gate_bias(bfs, bfl, bal, bo, bm, col0);
    float bb1 = gate_bias(bfs, bfl, bal, bo, bm, col1);

    if (blockIdx.x < 32) {
        // f_short / f_long / alpha / o columns -> global sg
#pragma unroll
        for (int r = 0; r < 8; ++r) {
            int row = rowbase + r;
            sg[row * NSG + col0] = sigmoidf_(acc0[r] + bb0);
            sg[row * NSG + col1] = sigmoidf_(acc1[r] + bb1);
        }
    } else {
        // m columns: stage sigmoid(m) in LDS, then emit m*x A-fragments
        int dl0 = nl * 16 + (lane & 15);        // d_local of tile0 column
#pragma unroll
        for (int r = 0; r < 8; ++r) {
            int row = rowbase + r;
            mlds[row][dl0]      = sigmoidf_(acc0[r] + bb0);
            mlds[row][dl0 + 16] = sigmoidf_(acc1[r] + bb1);
        }
        __syncthreads();
        int bp    = blockIdx.x - 32;            // 0..3
        int tid   = threadIdx.x;
        int pmt   = tid >> 6;                   // mtile 0..3
        int ksl   = (tid >> 5) & 1;             // 0..1
        int plane = tid & 31;
        int prow  = pmt * 16 + (plane & 15);
        int pkh   = (plane >> 4) ? 8 : 0;
        int ks_g  = KS_H + 2 * bp + ksl;        // 16..23
        const float* xr = x + (size_t)prow * T_ * D_ + (size_t)t * D_;
        unsigned int* d = AmX + (((size_t)pmt * KS_X + (ks_g - KS_H)) * 32 + plane) * 8;
#pragma unroll
        for (int v = 0; v < 8; ++v) {
            int k  = ks_g * 32 + aoff(v) + pkh;
            int dd = k - U_;                    // 0..255
            int dlo = dd - 64 * bp;             // 0..62 within this block's LDS tile
            d[v] = pack2(mlds[prow][dlo] * xr[dd], mlds[prow][dlo + 1] * xr[dd + 1]);
        }
    }
}

// ---------------------------------------------------------------------------
// GEMM cell: [64,768]x[768,512] + cell update. Writes c_t (cbuf), out[:,t,:],
// and packs h_t into the NEXT step's Ah fragments via LDS (double-buffered).
// ---------------------------------------------------------------------------
__global__ void gemm_cell(const unsigned int* __restrict__ Ah,
                          const unsigned int* __restrict__ AmX,
                          const unsigned int* __restrict__ Bc,
                          const float* __restrict__ bC, const float* __restrict__ sg,
                          float* __restrict__ cbuf,
                          unsigned int* __restrict__ AhNext,
                          float* __restrict__ out, int t) {
    __shared__ float hlds[64][64];
    int lane   = threadIdx.x & 31;
    int wave   = threadIdx.x >> 5;
    int mtile  = wave & 3;
    int nl     = (wave >> 2) * 2;
    int ntile0 = blockIdx.x * 4 + nl;
    int ntile1 = ntile0 + 1;

    const unsigned int* ahp = Ah  + (((size_t)mtile * KS_H) * 32 + lane) * 8;
    const unsigned int* axp = AmX + (((size_t)mtile * KS_X) * 32 + lane) * 8;
    const unsigned int* bp0 = Bc + (((size_t)ntile0 * KS) * 32 + lane) * 8;
    const unsigned int* bp1 = Bc + (((size_t)ntile1 * KS) * 32 + lane) * 8;

    v8f acc0 = {}; v8f acc1 = {};
    mma_seg(ahp, bp0, bp1, KS_H, acc0, acc1);   // h part
    mma_seg(axp, bp0, bp1, KS_X, acc0, acc1);   // m*x part

    int u0      = ntile0 * 16 + (lane & 15);
    int u1      = ntile1 * 16 + (lane & 15);
    int rowbase = mtile * 16 + ((lane >> 4) << 3);
    int ul0     = nl * 16 + (lane & 15);        // u_local in block's 64-wide tile
    float bc0 = bC[u0], bc1 = bC[u1];
#pragma unroll
    for (int r = 0; r < 8; ++r) {
        int row = rowbase + r;
#pragma unroll
        for (int s = 0; s < 2; ++s) {
            int   u    = s ? u1 : u0;
            int   ul   = s ? (ul0 + 16) : ul0;
            float pre  = (s ? acc1[r] + bc1 : acc0[r] + bc0);
            float chat = tanhf(pre);
            float fs = sg[row * NSG + u];
            float fl = sg[row * NSG + 512 + u];
            float al = sg[row * NSG + 1024 + u];
            float og = sg[row * NSG + 1536 + u];
            float f  = al * fs + (1.0f - al) * fl;
            float c  = f * cbuf[row * U_ + u] + (1.0f - f) * chat;
            cbuf[row * U_ + u] = c;
            float h = og * tanhf(c);
            hlds[row][ul] = h;
            out[((size_t)row * T_ + t) * U_ + u] = h;
        }
    }
    __syncthreads();
    // Emit next-step h A-fragments: this block owns ksteps 2b, 2b+1
    int tid   = threadIdx.x;
    int pmt   = tid >> 6;
    int ksl   = (tid >> 5) & 1;
    int plane = tid & 31;
    int prow  = pmt * 16 + (plane & 15);
    int pkh   = (plane >> 4) ? 8 : 0;
    int ks_g  = 2 * blockIdx.x + ksl;           // 0..15
    unsigned int* d = AhNext + (((size_t)pmt * KS_H + ks_g) * 32 + plane) * 8;
#pragma unroll
    for (int v = 0; v < 8; ++v) {
        int k  = ks_g * 32 + aoff(v) + pkh;     // == u (global unit index)
        int ulo = k - 64 * blockIdx.x;          // 0..62
        d[v] = pack2(hlds[prow][ulo], hlds[prow][ulo + 1]);
    }
}

// ---------------------------------------------------------------------------
// Host-side launch chain (graph-capture safe; all on `stream`).
// ---------------------------------------------------------------------------
extern "C" void kernel_launch(void* const* d_in, const int* in_sizes, int n_in,
                              void* d_out, int out_size, void* d_ws, size_t ws_size,
                              hipStream_t stream) {
    const float* x   = (const float*)d_in[0];
    const float* h0  = (const float*)d_in[1];
    const float* c0  = (const float*)d_in[2];
    const float* Wfs = (const float*)d_in[3];
    const float* bfs = (const float*)d_in[4];
    const float* Wfl = (const float*)d_in[5];
    const float* bfl = (const float*)d_in[6];
    const float* Wal = (const float*)d_in[7];
    const float* bal = (const float*)d_in[8];
    const float* Wm  = (const float*)d_in[9];
    const float* bm  = (const float*)d_in[10];
    const float* WC  = (const float*)d_in[11];
    const float* bC  = (const float*)d_in[12];
    const float* Wo  = (const float*)d_in[13];
    const float* bo  = (const float*)d_in[14];
    float* out = (float*)d_out;

    // Workspace carve-up (~21.9 MB)
    char* w = (char*)d_ws;
    unsigned int* Bg  = (unsigned int*)w; w += (size_t)NT_G * KS * 32 * 8 * 4;   // 3.54 MB
    unsigned int* Bc  = (unsigned int*)w; w += (size_t)NT_C * KS * 32 * 8 * 4;   // 0.79 MB
    unsigned int* AhA = (unsigned int*)w; w += (size_t)4 * KS_H * 32 * 8 * 4;    // 64 KB
    unsigned int* AhB = (unsigned int*)w; w += (size_t)4 * KS_H * 32 * 8 * 4;    // 64 KB
    unsigned int* AmX = (unsigned int*)w; w += (size_t)AX_STEP * 4;              // 32 KB
    unsigned int* AxX = (unsigned int*)w; w += (size_t)T_ * AX_STEP * 4;         // 16 MB
    float* sg   = (float*)w;              w += (size_t)B_ * NSG * 4;             // 512 KB
    float* cbuf = (float*)w;              w += (size_t)B_ * U_ * 4;              // 128 KB

    init_state<<<(B_ * U_ + 255) / 256, 256, 0, stream>>>(h0, c0, cbuf, AhA);
    pack_x_all<<<(T_ * 4 * KS_X * 32 + 255) / 256, 256, 0, stream>>>(x, AxX);
    {
        const int tot = (NT_G + NT_C) * KS * 32;
        pack_weights<<<(tot + 255) / 256, 256, 0, stream>>>(Wfs, Wfl, Wal, Wo, Wm, WC, Bg, Bc);
    }

    for (int t = 0; t < T_; ++t) {
        unsigned int* AhCur = (t & 1) ? AhB : AhA;
        unsigned int* AhNxt = (t & 1) ? AhA : AhB;
        gemm_gates<<<36, 256, 0, stream>>>(AhCur, AxX + (size_t)t * AX_STEP, Bg,
                                           bfs, bfl, bal, bo, bm, sg, x, AmX, t);
        gemm_cell<<<8, 256, 0, stream>>>(AhCur, AmX, Bc, bC, sg, cbuf, AhNxt, out, t);
    }
}
